// GAT_33217277067783
// MI455X (gfx1250) — matrix operand order repaired and checked by reading, not compiled, not run on hardware
//
#include <hip/hip_runtime.h>
#include <hip/hip_bf16.h>

// GAT forward, MI455X (gfx1250, wave32, WMMA).
// Strategy: never materialize the (B,H,S,S) score tensor (537MB). Rank-1
// logits + monotone leaky-relu => exact row-max via a cheap s2 scan, then a
// single exp pass feeding v_wmma_f32_16x16x32_bf16 for attn @ P.
// Projection also runs on bf16 WMMA. Accumulation is f32 throughout.

#define NEGV (-1000000000.0f)
#define LNEPS 1e-5f

constexpr int Bc = 16, Sc = 1024, Mc = 256, Hc = 8, Lc = 2, DKc = 32;

typedef __attribute__((ext_vector_type(16))) __bf16 v16bf;
typedef __attribute__((ext_vector_type(8)))  __bf16 v8bf;
typedef __attribute__((ext_vector_type(8)))  float  v8f;

static __device__ __forceinline__ float wxor(float v, int m) {
  return __shfl_xor(v, m, 32);
}

// ---------------- weight prep: Wt[l][n=h*32+d][m] = bf16(proj_w[l,h,m,d]) ---
__global__ void k_prep_w(const float* __restrict__ pw, __bf16* __restrict__ Wt) {
  int tid = blockIdx.x * 256 + threadIdx.x;      // L*256*256 = 131072
  int m = tid & 255;
  int n = (tid >> 8) & 255;
  int l = tid >> 16;
  int h = n >> 5, d = n & 31;
  Wt[tid] = (__bf16)pw[((l * Hc + h) * Mc + m) * DKc + d];
}

// ---------------- projection: Pt[b,h,d,j] = bf16( x @ W + b ) ---------------
// grid = B*S/16 row tiles, 256 threads = 8 waves, wave w == head w (DK==32).
__global__ void __launch_bounds__(256) k_proj(
    const float* __restrict__ x,      // (B*S, M) fp32
    const __bf16* __restrict__ Wt,    // (256, 256) this layer, [n][m]
    const float* __restrict__ pb,     // (H, DK) this layer
    __bf16* __restrict__ Pt)          // (B, H, DK, S)
{
  __shared__ __align__(16) __bf16 xs[16 * Mc];   // 8 KB
  const int tile = blockIdx.x;
  const int tid  = threadIdx.x;
  for (int k = 0; k < 16; ++k) {                 // stage x tile as bf16
    int flat = tid + k * 256;
    xs[flat] = (__bf16)x[(size_t)(tile * 16 + (flat >> 8)) * Mc + (flat & 255)];
  }
  __syncthreads();

  const int w  = tid >> 5;           // head
  const int l  = tid & 31;
  const int nl = l & 15;
  const int hi = l >> 4;             // lane half
  v8f c0 = {}, c1 = {};
  const __bf16* wrow0 = Wt + (size_t)(w * 32 + nl) * Mc;
  const __bf16* wrow1 = Wt + (size_t)(w * 32 + 16 + nl) * Mc;
  const __bf16* arow  = xs + (size_t)nl * Mc;

  for (int kc = 0; kc < Mc; kc += 32) {
    // A frag: row = nl, K = hi*16+e (contiguous, matches B)
    v8bf alo = *(const v8bf*)&arow[kc + hi * 16];
    v8bf ahi = *(const v8bf*)&arow[kc + hi * 16 + 8];
    v16bf a = __builtin_shufflevector(alo, ahi,
        0,1,2,3,4,5,6,7,8,9,10,11,12,13,14,15);
    // B frag: lane = N column, 16 contiguous K values (32B load)
    v16bf b0 = *(const v16bf*)&wrow0[kc + hi * 16];
    v16bf b1 = *(const v16bf*)&wrow1[kc + hi * 16];
    c0 = __builtin_amdgcn_wmma_f32_16x16x32_bf16(false, a, false, b0, (short)0, c0, false, false);
    c1 = __builtin_amdgcn_wmma_f32_16x16x32_bf16(false, a, false, b1, (short)0, c1, false, false);
  }
  // epilogue: bias, store transposed bf16 P (rows of C are contiguous j)
  const int b  = tile >> 6;                 // 64 tiles per batch
  const int j0 = (tile & 63) * 16;
  const float bias0 = pb[w * DKc + nl];
  const float bias1 = pb[w * DKc + 16 + nl];
  v8bf o0, o1;
  for (int r = 0; r < 8; ++r) {
    o0[r] = (__bf16)(c0[r] + bias0);
    o1[r] = (__bf16)(c1[r] + bias1);
  }
  __bf16* base = Pt + (size_t)(b * Hc + w) * DKc * Sc;
  *(v8bf*)&base[(size_t)nl        * Sc + j0 + hi * 8] = o0;
  *(v8bf*)&base[(size_t)(16 + nl) * Sc + j0 + hi * 8] = o1;
}

// ---------------- s1/s2 = P . a1/a2 ----------------------------------------
__global__ void k_scores(const __bf16* __restrict__ Pt, const float* __restrict__ aw,
                         float* __restrict__ s1, float* __restrict__ s2) {
  int tid = blockIdx.x * 256 + threadIdx.x;     // B*H*S
  int j  = tid & (Sc - 1);
  int bh = tid >> 10;
  const __bf16* p = Pt + (size_t)bh * DKc * Sc + j;
  float a = 0.f, b2 = 0.f;
  for (int d = 0; d < DKc; ++d) {
    float v = (float)p[(size_t)d * Sc];
    a  = fmaf(v, aw[d], a);
    b2 = fmaf(v, aw[DKc + d], b2);
  }
  s1[tid] = a; s2[tid] = b2;
}

// ---------------- fused attention + residual + layernorm --------------------
// grid = (b, 16-row tile), 256 threads = 8 waves, wave w == head w.
__global__ void __launch_bounds__(256) k_attn(
    const __bf16* __restrict__ Pt,
    const float* __restrict__ s1g, const float* __restrict__ s2g,
    const int* __restrict__ adj, const unsigned char* __restrict__ smask,
    const int* __restrict__ tptr, int layer, const float* __restrict__ abp,
    const float* __restrict__ xsrc,
    const float* __restrict__ lng, const float* __restrict__ lnb,
    float* __restrict__ dst)
{
  __shared__ float s2s[Hc * Sc];            // 32 KB
  __shared__ unsigned int mbits[16 * 32];   //  2 KB (bit-packed mask tile)
  __shared__ float rbuf[16 * Mc];           // 16 KB
  __shared__ float lsum[Hc * 16];

  const int blk = blockIdx.x;
  const int b   = blk >> 6;
  const int i0  = (blk & 63) * 16;
  const int tid = threadIdx.x;
  const bool use_adj = (layer == 0) || (tptr[0] != 1);
  const float ab = abp[0];

  for (int k = tid; k < Hc * Sc; k += 256)           // stage s2 (all heads)
    s2s[k] = s2g[(size_t)b * Hc * Sc + k];
  for (int wd = tid; wd < 512; wd += 256) {          // stage mask bits
    int row = wd >> 5, wj = wd & 31;
    size_t base = ((size_t)b * Sc + i0 + row) * Sc + (size_t)wj * 32;
    unsigned int bits = 0u;
    for (int k = 0; k < 32; ++k) {
      bool msk = (smask[base + k] != 0) || (use_adj && adj[base + k] == 0);
      bits |= (unsigned int)msk << k;
    }
    mbits[wd] = bits;
  }
  __syncthreads();

  const int w   = tid >> 5;
  const int l   = tid & 31;
  const int row = l & 15;
  const int hi  = l >> 4;
  const int nl  = row;

  const float s1v = s1g[(size_t)(b * Hc + w) * Sc + i0 + row] + ab;
  const float* s2w = s2s + w * Sc;

  // pass 1: row max = leaky(s1+ab + max unmasked s2)  (leaky is monotone)
  float mx = -3.0e38f;
  for (int wj = hi * 16; wj < hi * 16 + 16; ++wj) {
    unsigned int bits = mbits[row * 32 + wj];
    for (int k = 0; k < 32; ++k)
      if (!((bits >> k) & 1u)) mx = fmaxf(mx, s2w[wj * 32 + k]);
  }
  mx = fmaxf(mx, wxor(mx, 16));
  float t = s1v + mx;
  float mrow = (t >= 0.f) ? t : 0.2f * t;
  if (mx < -1.0e37f) mrow = NEGV;   // all masked -> uniform softmax

  // pass 2: p = exp(score - m) as bf16 A frag; C += A @ Pt
  v8f c0 = {}, c1 = {};
  float psum = 0.f;
  const __bf16* pbase = Pt + (size_t)(b * Hc + w) * DKc * Sc;
  for (int jc = 0; jc < Sc; jc += 32) {
    unsigned int bits = mbits[row * 32 + (jc >> 5)];
    v16bf a;
    for (int e = 0; e < 16; ++e) {
      int jo = hi * 16 + e;
      float sc;
      if ((bits >> jo) & 1u) sc = NEGV;
      else { float u = s1v + s2w[jc + jo]; sc = (u >= 0.f) ? u : 0.2f * u; }
      float p = __expf(sc - mrow);
      psum += p;
      a[e] = (__bf16)p;
    }
    v16bf b0 = *(const v16bf*)&pbase[(size_t)nl        * Sc + jc + hi * 16];
    v16bf b1 = *(const v16bf*)&pbase[(size_t)(16 + nl) * Sc + jc + hi * 16];
    c0 = __builtin_amdgcn_wmma_f32_16x16x32_bf16(false, a, false, b0, (short)0, c0, false, false);
    c1 = __builtin_amdgcn_wmma_f32_16x16x32_bf16(false, a, false, b1, (short)0, c1, false, false);
  }
  psum += wxor(psum, 16);
  if (l < 16) lsum[w * 16 + row] = psum;
  __syncthreads();

  // normalize + residual into LDS (concat: head w -> cols w*32..w*32+31)
  const size_t xrow = (size_t)b * Sc + i0;
  for (int r = 0; r < 8; ++r) {
    int ir = r + hi * 8;
    float inv = 1.0f / lsum[w * 16 + ir];
    int col = w * 32 + nl;
    rbuf[ir * Mc + col]      = c0[r] * inv + xsrc[(xrow + ir) * Mc + col];
    rbuf[ir * Mc + col + 16] = c1[r] * inv + xsrc[(xrow + ir) * Mc + col + 16];
  }
  __syncthreads();

  // layernorm: wave w handles rows 2w, 2w+1; wave32 shuffle reduction
  for (int rr = 2 * w; rr <= 2 * w + 1; ++rr) {
    float s = 0.f;
    for (int q = 0; q < 8; ++q) s += rbuf[rr * Mc + q * 32 + l];
    s += wxor(s, 1); s += wxor(s, 2); s += wxor(s, 4); s += wxor(s, 8); s += wxor(s, 16);
    float mu = s * (1.0f / Mc);
    float v2 = 0.f;
    for (int q = 0; q < 8; ++q) { float d2 = rbuf[rr * Mc + q * 32 + l] - mu; v2 += d2 * d2; }
    v2 += wxor(v2, 1); v2 += wxor(v2, 2); v2 += wxor(v2, 4); v2 += wxor(v2, 8); v2 += wxor(v2, 16);
    float rstd = rsqrtf(v2 * (1.0f / Mc) + LNEPS);
    for (int q = 0; q < 8; ++q) {
      int m = q * 32 + l;
      dst[(xrow + rr) * Mc + m] = (rbuf[rr * Mc + m] - mu) * rstd * lng[m] + lnb[m];
    }
  }
}

// ---------------------------------------------------------------------------
extern "C" void kernel_launch(void* const* d_in, const int* in_sizes, int n_in,
                              void* d_out, int out_size, void* d_ws, size_t ws_size,
                              hipStream_t stream) {
  const int*           adj    = (const int*)d_in[0];
  const float*         x0     = (const float*)d_in[1];
  const unsigned char* smask  = (const unsigned char*)d_in[2];
  const int*           tptr   = (const int*)d_in[3];
  const float*         pw     = (const float*)d_in[4];
  const float*         pb     = (const float*)d_in[5];
  const float*         aw     = (const float*)d_in[6];
  const float*         abv    = (const float*)d_in[7];
  const float*         lng    = (const float*)d_in[8];
  const float*         lnb    = (const float*)d_in[9];
  float*               out    = (float*)d_out;

  char* ws = (char*)d_ws;
  size_t off = 0;
  auto alloc = [&](size_t bytes) { char* p = ws + off; off = (off + bytes + 255) & ~(size_t)255; return p; };
  __bf16* Wt  = (__bf16*)alloc((size_t)Lc * 256 * 256 * 2);          // 256 KB
  __bf16* Pt  = (__bf16*)alloc((size_t)Bc * Hc * DKc * Sc * 2);      //   8 MB
  float*  s1  = (float*) alloc((size_t)Bc * Hc * Sc * 4);            // 512 KB
  float*  s2  = (float*) alloc((size_t)Bc * Hc * Sc * 4);            // 512 KB
  float*  x1  = (float*) alloc((size_t)Bc * Sc * Mc * 4);            //  16 MB

  k_prep_w<<<(Lc * 256 * 256) / 256, 256, 0, stream>>>(pw, Wt);

  for (int layer = 0; layer < Lc; ++layer) {
    const float* xsrc = (layer == 0) ? x0 : x1;
    float*       dst  = (layer == Lc - 1) ? out : x1;
    k_proj  <<<Bc * Sc / 16, 256, 0, stream>>>(
        xsrc, Wt + (size_t)layer * 256 * 256, pb + (size_t)layer * Hc * DKc, Pt);
    k_scores<<<(Bc * Hc * Sc) / 256, 256, 0, stream>>>(
        Pt, aw + (size_t)layer * 2 * DKc, s1, s2);
    k_attn  <<<Bc * Sc / 16, 256, 0, stream>>>(
        Pt, s1, s2, adj, smask, tptr, layer, abv + layer,
        xsrc, lng + (size_t)layer * Mc, lnb + (size_t)layer * Mc, dst);
  }
}
